// Agent_16904991277190
// MI455X (gfx1250) — matrix operand order repaired and checked by reading, not compiled, not run on hardware
//
#include <hip/hip_runtime.h>
#include <cstdint>
#include <cstddef>

// ---------------------------------------------------------------------------
// PPO loss for MI455X (gfx1250, wave32, WMMA).
// Heavy work: two 3-layer MLPs -> f16 WMMA GEMMs (v_wmma_f32_16x16x32_f16)
// with async global->LDS staging (ASYNCcnt) and double-buffered LDS tiles.
// BN=128 for the H=512 layers (wave tile 32x64 = 8 wmma/K-step/wave),
// BN=64 for the narrow policy head (N=64).
// ---------------------------------------------------------------------------

namespace {
constexpr int T_DIM   = 128;
constexpr int B_DIM   = 1024;
constexpr int OBS_DIM = 256;
constexpr int ACT_DIM = 32;
constexpr int H_DIM   = 512;
constexpr int M_POL   = T_DIM * B_DIM;        // 131072
constexpr int M_VAL   = (T_DIM + 1) * B_DIM;  // 132096

constexpr float DISCOUNT     = 0.97f;
constexpr float GAE_LAMBDA   = 0.95f;
constexpr float RSCALE       = 0.1f;
constexpr float CLIP_EPS     = 0.3f;
constexpr float ENT_COST     = 0.01f;
constexpr float HALF_LOG_2PI = 0.91893853320467274178f;
constexpr float LOG2_C       = 0.69314718055994530942f;
}

typedef __attribute__((ext_vector_type(16))) _Float16 v16h;
typedef __attribute__((ext_vector_type(8)))  _Float16 v8h;
typedef __attribute__((ext_vector_type(8)))  float    v8f;

__device__ __forceinline__ float fast_rcp(float x) {
  return __builtin_amdgcn_rcpf(x);
}
__device__ __forceinline__ float softplusf(float x) {
  return fmaxf(x, 0.0f) + log1pf(__expf(-fabsf(x)));
}
__device__ __forceinline__ float siluf(float x) {
  return x * fast_rcp(1.0f + __expf(-x));
}
__device__ __forceinline__ uint32_t hashu(uint32_t x) {
  x ^= x >> 17; x *= 0xed5ad4bbu;
  x ^= x >> 11; x *= 0xac4c1b51u;
  x ^= x >> 15; x *= 0x31848babu;
  x ^= x >> 14;
  return x;
}
__device__ __forceinline__ v16h cat8(v8h lo, v8h hi) {
  v16h r;
#pragma unroll
  for (int j = 0; j < 8; ++j) { r[j] = lo[j]; r[j + 8] = hi[j]; }
  return r;
}

// --- CDNA5 async tensor path: global -> LDS, tracked by ASYNCcnt -----------
__device__ __forceinline__ void async_copy_b128(const _Float16* __restrict__ g,
                                                _Float16* l) {
  uint32_t lds_addr = (uint32_t)(uintptr_t)l;          // LDS address is the low 32 bits of the pointer
  uint64_t gaddr    = (uint64_t)(uintptr_t)g;
  asm volatile("global_load_async_to_lds_b128 %0, %1, off"
               :: "v"(lds_addr), "v"(gaddr)
               : "memory");
}
__device__ __forceinline__ void wait_async_le2() {
  asm volatile("s_wait_asynccnt 0x2" ::: "memory");
}
__device__ __forceinline__ void wait_async_0() {
  asm volatile("s_wait_asynccnt 0x0" ::: "memory");
}

// ---------------------------------------------------------------------------
// Elementwise kernels
// ---------------------------------------------------------------------------
__global__ void Agent_norm_obs(const float* __restrict__ obs,
                               const float* __restrict__ mean,
                               const float* __restrict__ var,
                               const int*   __restrict__ nsp,
                               _Float16* __restrict__ out, int total) {
  int i = blockIdx.x * blockDim.x + threadIdx.x;
  if (i >= total) return;
  int c = i & (OBS_DIM - 1);
  float ns = (float)nsp[0];
  float v  = var[c] / (ns + 1.0f);
  v = fminf(fmaxf(v, 1e-6f), 1e6f);
  float x = (obs[i] - mean[c]) * rsqrtf(v);
  x = fminf(fmaxf(x, -5.0f), 5.0f);
  out[i] = (_Float16)x;
}

__global__ void Agent_f32_to_f16(const float* __restrict__ in,
                                 _Float16* __restrict__ out, int n) {
  int i = blockIdx.x * blockDim.x + threadIdx.x;
  if (i < n) out[i] = (_Float16)in[i];
}

__global__ void Agent_zero_acc(float* acc) {
  if (threadIdx.x < 4) acc[threadIdx.x] = 0.0f;
}

// ---------------------------------------------------------------------------
// WMMA GEMM:  C[M,N] = act(A[M,K] @ W[K,N] + bias)
//   A, W f16 row-major.  BM=128, BK=32, BN template (64 or 128).
//   256 threads = 8 waves (4 in M x 2 in N); wave tile = 32 x (BN/2)
//   -> 2 x (BN/32) v_wmma_f32_16x16x32_f16 accumulators per wave.
//   A tile staged global->LDS via global_load_async_to_lds_b128 (2-deep
//   pipeline, ASYNCcnt); W tile staged via registers with LDS transpose.
//   Requires M%128==0, N%BN==0, K%32==0 (true for all calls here).
// ---------------------------------------------------------------------------
template <bool SILU, bool OUT_F16, int BN>
__global__ __launch_bounds__(256) void Agent_gemm_wmma(
    const _Float16* __restrict__ A, const _Float16* __restrict__ W,
    const float* __restrict__ bias, void* __restrict__ Cout,
    int M, int N, int K) {
  constexpr int BM = 128, BK = 32;
  constexpr int WN   = BN / 2;    // per-wave N span (32 or 64)
  constexpr int NT   = WN / 16;   // 16-wide n-tiles per wave (2 or 4)
  constexpr int CH   = BN / 8;    // W-staging halves per thread (8 or 16)
  constexpr int WREG = CH / 8;    // v8h staging regs per thread (1 or 2)

  __shared__ __align__(16) _Float16 As[2][BM][BK];   // 2 x 8 KB  (m, k)
  __shared__ __align__(16) _Float16 Bs[2][BN][BK];   // 2 x BN/16 KB (n, k) transposed

  const int tid  = threadIdx.x;
  const int bm   = blockIdx.x * BM;
  const int bn   = blockIdx.y * BN;
  const int wave = tid >> 5;
  const int lane = tid & 31;
  const int wm   = wave & 3;   // 0..3  (M direction, 32 rows each)
  const int wn   = wave >> 2;  // 0..1  (N direction, WN cols each)
  const int rlo  = lane & 15;
  const int kg   = lane >> 4;  // K half-group per ISA A/B layouts

  // A async staging: 512 x 16B chunks per tile; this thread owns chunks
  // (tid) and (tid+256) -> 2 async b128 issues per wave per tile.
  const int c0r = tid >> 2,          c0s = (tid & 3) * 8;
  const int c1r = (tid + 256) >> 2,  c1s = ((tid + 256) & 3) * 8;

  // W register staging: CH halves of one k-row per thread (8 threads/row).
  const int wk  = tid >> 3;          // 0..31
  const int wn0 = (tid & 7) * CH;

  const int steps = K / BK;
  v8f acc[2][NT] = {};
  v8h wreg[WREG];

  auto issueA = [&](int s, int p) {
    const _Float16* base = A + (size_t)bm * K + s * BK;
    async_copy_b128(base + (size_t)c0r * K + c0s, &As[p][c0r][c0s]);
    async_copy_b128(base + (size_t)c1r * K + c1s, &As[p][c1r][c1s]);
  };
  auto loadW = [&](int s) {
#pragma unroll
    for (int r = 0; r < WREG; ++r)
      wreg[r] = *(const v8h*)(W + (size_t)(s * BK + wk) * N + (bn + wn0 + r * 8));
  };
  auto storeW = [&](int p) {
#pragma unroll
    for (int r = 0; r < WREG; ++r)
#pragma unroll
      for (int j = 0; j < 8; ++j) Bs[p][wn0 + r * 8 + j][wk] = wreg[r][j];
  };

  loadW(0);
  issueA(0, 0);

  for (int s = 0; s < steps; ++s) {
    const int p = s & 1;
    storeW(p);                       // W tile s -> LDS (transposed)
    if (s + 1 < steps) {
      issueA(s + 1, p ^ 1);          // prefetch next A tile (other buffer)
      wait_async_le2();              // tile s's asyncs complete (in-order)
      loadW(s + 1);                  // overlap next W fetch with compute
    } else {
      wait_async_0();
    }
    __syncthreads();                 // all waves' A(s)/W(s) visible

    // ---- fragments per ISA 16-bit A(16x32)/B(32x16) layouts ----
    v16h af[2], bf[NT];
#pragma unroll
    for (int im = 0; im < 2; ++im) {
      int r = wm * 32 + im * 16 + rlo;
      v8h lo = *(const v8h*)&As[p][r][kg * 8];
      v8h hi = *(const v8h*)&As[p][r][16 + kg * 8];
      af[im] = cat8(lo, hi);
    }
#pragma unroll
    for (int in = 0; in < NT; ++in) {
      int c = wn * WN + in * 16 + rlo;
      v8h lo = *(const v8h*)&Bs[p][c][kg * 8];
      v8h hi = *(const v8h*)&Bs[p][c][16 + kg * 8];
      bf[in] = cat8(lo, hi);
    }

#pragma unroll
    for (int im = 0; im < 2; ++im)
#pragma unroll
      for (int in = 0; in < NT; ++in)
        acc[im][in] = __builtin_amdgcn_wmma_f32_16x16x32_f16(
            false, af[im], false, bf[in], (short)0, acc[im][in], false, false);

    __syncthreads();                 // buffers free for next prefetch wave
  }

  // ---- epilogue: C 16x16 f32 layout (lane -> N, vgpr -> M) ----
  const int mofs = (lane >> 4) * 8;
#pragma unroll
  for (int im = 0; im < 2; ++im) {
#pragma unroll
    for (int in = 0; in < NT; ++in) {
      int row0 = bm + wm * 32 + im * 16 + mofs;
      int col  = bn + wn * WN + in * 16 + rlo;
      float bv = bias ? bias[col] : 0.0f;
#pragma unroll
      for (int v = 0; v < 8; ++v) {
        float x = acc[im][in][v] + bv;
        if (SILU) x = siluf(x);
        size_t idx = (size_t)(row0 + v) * N + col;
        if (OUT_F16) ((_Float16*)Cout)[idx] = (_Float16)x;
        else         ((float*)Cout)[idx]    = x;
      }
    }
  }
}

// ---------------------------------------------------------------------------
// Value head: baseline[r] = h2[r,:] . vw2 + vb2   (N == 1)
// ---------------------------------------------------------------------------
__global__ void Agent_rowdot(const _Float16* __restrict__ h,
                             const _Float16* __restrict__ w,
                             const float* __restrict__ b,
                             float* __restrict__ out, int M, int K) {
  int r = blockIdx.x * blockDim.x + threadIdx.x;
  if (r >= M) return;
  const v8h* hp = (const v8h*)(h + (size_t)r * K);
  const v8h* wp = (const v8h*)w;
  float s = 0.0f;
  for (int k = 0; k < K / 8; ++k) {
    v8h hv = hp[k], wv = wp[k];
#pragma unroll
    for (int j = 0; j < 8; ++j) s += (float)hv[j] * (float)wv[j];
  }
  out[r] = s + b[0];
}

// ---------------------------------------------------------------------------
// GAE: reverse scan over T (one thread per b), then forward advantage pass.
// ---------------------------------------------------------------------------
__global__ __launch_bounds__(1024) void Agent_gae(
    const float* __restrict__ reward, const float* __restrict__ done,
    const float* __restrict__ trunc,  const float* __restrict__ baseline,
    float* __restrict__ vs, float* __restrict__ adv) {
  int b = threadIdx.x;
  float boot = baseline[(size_t)T_DIM * B_DIM + b];
  float acc = 0.0f;
  for (int t = T_DIM - 1; t >= 0; --t) {
    int i = t * B_DIM + b;
    float tr   = trunc[i];
    float term = done[i] * (1.0f - tr);
    float tm   = 1.0f - tr;
    float vt   = baseline[i];
    float vtp1 = (t == T_DIM - 1) ? boot : baseline[i + B_DIM];
    float r    = reward[i] * RSCALE;
    float nd   = DISCOUNT * (1.0f - term);
    float delta = (r + nd * vtp1 - vt) * tm;
    acc = delta + nd * tm * GAE_LAMBDA * acc;
    vs[i] = acc + vt;
  }
  for (int t = 0; t < T_DIM; ++t) {
    int i = t * B_DIM + b;
    float tr    = trunc[i];
    float term  = done[i] * (1.0f - tr);
    float tm    = 1.0f - tr;
    float vstp1 = (t == T_DIM - 1) ? boot : vs[i + B_DIM];
    adv[i] = (reward[i] * RSCALE + DISCOUNT * (1.0f - term) * vstp1
              - baseline[i]) * tm;
  }
}

// ---------------------------------------------------------------------------
// Per-sample losses (log-probs, entropy, surrogate) + block reduce + atomics.
// ---------------------------------------------------------------------------
__global__ __launch_bounds__(256) void Agent_loss(
    const float* __restrict__ plog, const float* __restrict__ blog,
    const float* __restrict__ act,  const float* __restrict__ adv,
    const float* __restrict__ vs,   const float* __restrict__ baseline,
    float* __restrict__ accs, int n) {
  int tid = threadIdx.x;
  int i   = blockIdx.x * 256 + tid;
  float pol = 0.0f, v2 = 0.0f, ent = 0.0f;
  if (i < n) {
    const float* pl = plog + (size_t)i * (2 * ACT_DIM);
    const float* bl = blog + (size_t)i * (2 * ACT_DIM);
    const float* a  = act  + (size_t)i * ACT_DIM;
    float tlp = 0.0f, blp = 0.0f, es = 0.0f;
#pragma unroll 4
    for (int j = 0; j < ACT_DIM; ++j) {
      float av    = a[j];
      float ldj_a = 2.0f * (LOG2_C - av - softplusf(-2.0f * av));
      {  // behaviour
        float loc = bl[j], sc = softplusf(bl[j + ACT_DIM]) + 0.001f;
        float z = (av - loc) * fast_rcp(sc);
        blp += -0.5f * z * z - (HALF_LOG_2PI + __logf(sc)) - ldj_a;
      }
      float loc = pl[j], sc = softplusf(pl[j + ACT_DIM]) + 0.001f;
      {  // target
        float z = (av - loc) * fast_rcp(sc);
        tlp += -0.5f * z * z - (HALF_LOG_2PI + __logf(sc)) - ldj_a;
      }
      {  // entropy sample (hash-based gaussian)
        uint32_t s  = (uint32_t)(i * ACT_DIM + j);
        float u1 = ((float)hashu(s) + 1.0f) * 2.3283064e-10f;
        float u2 = (float)hashu(s ^ 0x9e3779b9u) * 2.3283064e-10f;
        float eps  = sqrtf(-2.0f * __logf(u1)) * __cosf(6.2831853f * u2);
        float dist = loc + sc * eps;
        es += 0.5f + HALF_LOG_2PI + __logf(sc)
              + 2.0f * (LOG2_C - dist - softplusf(-2.0f * dist));
      }
    }
    float rho = __expf(tlp - blp);
    float rc  = fminf(fmaxf(rho, 1.0f - CLIP_EPS), 1.0f + CLIP_EPS);
    float ad  = adv[i];
    pol = fminf(rho * ad, rc * ad);
    float ve = vs[i] - baseline[i];
    v2  = ve * ve;
    ent = es;
  }
  __shared__ float sm0[256], sm1[256], sm2[256];
  sm0[tid] = pol; sm1[tid] = v2; sm2[tid] = ent;
  __syncthreads();
  for (int s = 128; s > 0; s >>= 1) {
    if (tid < s) {
      sm0[tid] += sm0[tid + s];
      sm1[tid] += sm1[tid + s];
      sm2[tid] += sm2[tid + s];
    }
    __syncthreads();
  }
  if (tid == 0) {
    atomicAdd(&accs[0], sm0[0]);
    atomicAdd(&accs[1], sm1[0]);
    atomicAdd(&accs[2], sm2[0]);
  }
}

__global__ void Agent_final(const float* __restrict__ accs,
                            float* __restrict__ out) {
  float n = (float)(T_DIM * B_DIM);
  float policy_loss  = -accs[0] / n;
  float v_loss       = (accs[1] / n) * 0.5f * 0.5f;
  float entropy_loss = -ENT_COST * (accs[2] / n);
  out[0] = policy_loss + v_loss + entropy_loss;
}

// ---------------------------------------------------------------------------
// Host side
// ---------------------------------------------------------------------------
extern "C" void kernel_launch(void* const* d_in, const int* in_sizes, int n_in,
                              void* d_out, int out_size, void* d_ws,
                              size_t ws_size, hipStream_t stream) {
  const float* observation = (const float*)d_in[0];
  const float* logits      = (const float*)d_in[1];
  const float* action      = (const float*)d_in[2];
  const float* reward      = (const float*)d_in[3];
  const float* done        = (const float*)d_in[4];
  const float* truncation  = (const float*)d_in[5];
  const float* rmean       = (const float*)d_in[6];
  const float* rvar        = (const float*)d_in[7];
  const float* pw0 = (const float*)d_in[8];   const float* pb0 = (const float*)d_in[9];
  const float* pw1 = (const float*)d_in[10];  const float* pb1 = (const float*)d_in[11];
  const float* pw2 = (const float*)d_in[12];  const float* pb2 = (const float*)d_in[13];
  const float* vw0 = (const float*)d_in[14];  const float* vb0 = (const float*)d_in[15];
  const float* vw1 = (const float*)d_in[16];  const float* vb1 = (const float*)d_in[17];
  const float* vw2 = (const float*)d_in[18];  const float* vb2 = (const float*)d_in[19];
  const int*   num_steps = (const int*)d_in[20];

  // ---- workspace layout ----
  char* ws = (char*)d_ws;
  size_t off = 0;
  auto carve = [&](size_t bytes) -> char* {
    char* p = ws + off;
    off = (off + bytes + 255) & ~(size_t)255;
    return p;
  };
  _Float16* obs16 = (_Float16*)carve((size_t)M_VAL * OBS_DIM * 2);
  _Float16* h1    = (_Float16*)carve((size_t)M_VAL * H_DIM * 2);
  _Float16* h2    = (_Float16*)carve((size_t)M_VAL * H_DIM * 2);
  float*    plog  = (float*)carve((size_t)M_POL * 2 * ACT_DIM * 4);
  float*    base  = (float*)carve((size_t)M_VAL * 4);
  float*    vsb   = (float*)carve((size_t)M_POL * 4);
  float*    advb  = (float*)carve((size_t)M_POL * 4);
  _Float16* pw0h  = (_Float16*)carve((size_t)OBS_DIM * H_DIM * 2);
  _Float16* pw1h  = (_Float16*)carve((size_t)H_DIM * H_DIM * 2);
  _Float16* pw2h  = (_Float16*)carve((size_t)H_DIM * 2 * ACT_DIM * 2);
  _Float16* vw0h  = (_Float16*)carve((size_t)OBS_DIM * H_DIM * 2);
  _Float16* vw1h  = (_Float16*)carve((size_t)H_DIM * H_DIM * 2);
  _Float16* vw2h  = (_Float16*)carve((size_t)H_DIM * 2);
  float*    accs  = (float*)carve(64);
  (void)ws_size; (void)in_sizes; (void)n_in; (void)out_size;

  // ---- weights -> f16 ----
  auto cvt = [&](const float* src, _Float16* dst, int n) {
    Agent_f32_to_f16<<<(n + 255) / 256, 256, 0, stream>>>(src, dst, n);
  };
  cvt(pw0, pw0h, OBS_DIM * H_DIM);
  cvt(pw1, pw1h, H_DIM * H_DIM);
  cvt(pw2, pw2h, H_DIM * 2 * ACT_DIM);
  cvt(vw0, vw0h, OBS_DIM * H_DIM);
  cvt(vw1, vw1h, H_DIM * H_DIM);
  cvt(vw2, vw2h, H_DIM);

  // ---- normalize observations -> f16 ----
  {
    int total = M_VAL * OBS_DIM;
    Agent_norm_obs<<<(total + 255) / 256, 256, 0, stream>>>(
        observation, rmean, rvar, num_steps, obs16, total);
  }

  dim3 blk(256);
  // ---- value net (all T+1 rows) ----
  Agent_gemm_wmma<true, true, 128><<<dim3(M_VAL / 128, H_DIM / 128), blk, 0, stream>>>(
      obs16, vw0h, vb0, h1, M_VAL, H_DIM, OBS_DIM);
  Agent_gemm_wmma<true, true, 128><<<dim3(M_VAL / 128, H_DIM / 128), blk, 0, stream>>>(
      h1, vw1h, vb1, h2, M_VAL, H_DIM, H_DIM);
  Agent_rowdot<<<(M_VAL + 255) / 256, 256, 0, stream>>>(
      h2, vw2h, vb2, base, M_VAL, H_DIM);

  // ---- policy net (first T rows), reuse h1/h2 ----
  Agent_gemm_wmma<true, true, 128><<<dim3(M_POL / 128, H_DIM / 128), blk, 0, stream>>>(
      obs16, pw0h, pb0, h1, M_POL, H_DIM, OBS_DIM);
  Agent_gemm_wmma<true, true, 128><<<dim3(M_POL / 128, H_DIM / 128), blk, 0, stream>>>(
      h1, pw1h, pb1, h2, M_POL, H_DIM, H_DIM);
  Agent_gemm_wmma<false, false, 64><<<dim3(M_POL / 128, (2 * ACT_DIM) / 64), blk, 0, stream>>>(
      h2, pw2h, pb2, plog, M_POL, 2 * ACT_DIM, H_DIM);

  // ---- GAE scan ----
  Agent_gae<<<1, B_DIM, 0, stream>>>(reward, done, truncation, base, vsb, advb);

  // ---- losses ----
  Agent_zero_acc<<<1, 32, 0, stream>>>(accs);
  Agent_loss<<<(M_POL + 255) / 256, 256, 0, stream>>>(
      plog, logits, action, advb, vsb, base, accs, M_POL);
  Agent_final<<<1, 1, 0, stream>>>(accs, (float*)d_out);
}
